// FizzBuzzTransformer_76209899700404
// MI455X (gfx1250) — matrix operand order repaired and checked by reading, not compiled, not run on hardware
//
#include <hip/hip_runtime.h>

// ---------------- problem constants (match reference) ----------------
#define B_   2
#define T_   1024
#define E_   1024
#define H_   16
#define D_   64
#define L_   8
#define V_   32000
#define F_   4096           // 4*E
#define NTOK (B_ * T_)      // 2048
#define SCALE_ 0.125f       // 1/sqrt(64)
#define EPS_   1e-5f

// GEMM tiling: block 128x128 (8 waves: 4 along M x 2 along N), wave 32x64, K slab 32.
#define BM_ 128
#define BN_ 128
#define BK_ 32
#define BKP_ 40   // padded LDS row (elements): 80B rows -> conflict-free ds_load_b128

typedef __bf16 bf16;
typedef __attribute__((ext_vector_type(8)))  __bf16 bf16x8;
typedef __attribute__((ext_vector_type(16))) __bf16 bf16x16;
typedef __attribute__((ext_vector_type(8)))  float  f32x8;

__device__ __forceinline__ bf16 cvt_bf16(float f) {
  union { float f; unsigned int u; } a; a.f = f;
  unsigned int lsb = (a.u >> 16) & 1u;
  unsigned short r = (unsigned short)((a.u + 0x7FFFu + lsb) >> 16);
  union { unsigned short s; bf16 h; } o; o.s = r;
  return o.h;
}

// ---------------- WMMA fragment loaders (CDNA5 16-bit layouts) ----------------
// A 16x32 (MxK): lanes 0-15 -> row M=lane, K {0..7, 16..23}; lanes 16-31 -> K {8..15, 24..31}
__device__ __forceinline__ bf16x16 load_frag_a(const bf16* base, int row0, int lda,
                                               int k0, int lane) {
  const bf16* p = base + (long long)(row0 + (lane & 15)) * lda + k0 + ((lane >> 4) << 3);
  union { bf16x16 v; bf16x8 h[2]; } u;
  u.h[0] = *(const bf16x8*)(p);
  u.h[1] = *(const bf16x8*)(p + 16);
  return u.v;
}
// B 32x16 (KxN) from LDS tile [col][k] (row stride BKP_): lanes 0-15 -> col N=lane,
// K {0..7, 16..23}; lanes 16-31 -> K {8..15, 24..31}.
__device__ __forceinline__ bf16x16 load_frag_b_lds(const bf16* tile, int coll, int lane) {
  const bf16* p = tile + (coll + (lane & 15)) * BKP_ + ((lane >> 4) << 3);
  union { bf16x16 v; bf16x8 h[2]; } u;
  u.h[0] = *(const bf16x8*)(p);
  u.h[1] = *(const bf16x8*)(p + 16);
  return u.v;
}

// ---------------- generic WMMA GEMM: C = scale*(A @ Bt^T) [+bias][+residual][ReLU] ------
// A:[M,K] bf16 (lda), Bt:[N,K] bf16 (ldb), C:[M,N] f32 or bf16 (ldc).
// z-offset: zb=z/zmod, zh=z%zmod; base += zb*s?0 + zh*s?1.
// M must be a multiple of 128; N a multiple of 32 (cols clamped on load, guarded on store).
__global__ __launch_bounds__(256) void gpt_wmma_gemm(
    const bf16* __restrict__ A, long long sA0, long long sA1,
    const bf16* __restrict__ Bt, long long sB0, long long sB1,
    float* __restrict__ outF, bf16* __restrict__ outB, long long sC0, long long sC1,
    int lda, int ldb, int ldc,
    const float* __restrict__ bias, const float* __restrict__ residual,
    int M, int N, int K, int zmod, float scale, int relu) {
  __shared__ bf16 ldsB[2][BN_ * BKP_];   // 2 x 10240 B, double buffered

  int z  = blockIdx.z;
  int zb = z / zmod, zh = z % zmod;
  A  += zb * sA0 + zh * sA1;
  Bt += zb * sB0 + zh * sB1;
  long long cbase = zb * sC0 + zh * sC1;

  int lane  = threadIdx.x & 31;
  int wave  = threadIdx.x >> 5;
  int mwave = wave >> 1;                       // 0..3
  int nwave = wave & 1;                        // 0..1
  int wm  = blockIdx.x * BM_ + mwave * 32;     // wave rows  [wm,  wm+32)
  int wn0 = blockIdx.y * BN_ + nwave * 64;     // wave cols  [wn0, wn0+64)

  // Cooperative async stage of Bt block slab [BN_ cols x BK_ K] into LDS.
  // tid -> col = tid>>1 (0..127), khalf = (tid&1)*16; two b128 copies per thread.
  int scol  = threadIdx.x >> 1;
  int khalf = (threadIdx.x & 1) << 4;
  int gcol  = blockIdx.y * BN_ + scol;
  if (gcol >= N) gcol = N - 1;                 // clamp (stores are guarded)
  const bf16* bsrc = Bt + (long long)gcol * ldb + khalf;
  unsigned lds_wr0 = (unsigned)(size_t)&ldsB[0][scol * BKP_ + khalf];
  unsigned lds_buf = (unsigned)((size_t)&ldsB[1][0] - (size_t)&ldsB[0][0]);

#define STAGE_B(k0_, buf_)                                                          \
  do {                                                                              \
    unsigned long long g_ = (unsigned long long)(size_t)(bsrc + (k0_));             \
    unsigned l_ = lds_wr0 + (unsigned)(buf_) * lds_buf;                             \
    asm volatile("global_load_async_to_lds_b128 %0, %1, off"                        \
                 :: "v"(l_), "v"(g_) : "memory");                                   \
    asm volatile("global_load_async_to_lds_b128 %0, %1, off"                        \
                 :: "v"(l_ + 16u), "v"(g_ + 16ull) : "memory");                     \
  } while (0)

  STAGE_B(0, 0);
  asm volatile("s_wait_asynccnt 0" ::: "memory");
  __syncthreads();

  f32x8 acc[2][4];
#pragma unroll
  for (int mi = 0; mi < 2; ++mi)
#pragma unroll
    for (int ni = 0; ni < 4; ++ni) acc[mi][ni] = f32x8{};

  int buf = 0;
  for (int k0 = 0; k0 < K; k0 += BK_) {
    if (k0 + BK_ < K) {
      STAGE_B(k0 + BK_, buf ^ 1);              // DMA next slab while computing
      __builtin_prefetch(A + (long long)(wm + (lane & 15)) * lda + k0 + BK_, 0, 1);
    }
    bf16x16 a0 = load_frag_a(A, wm,      lda, k0, lane);
    bf16x16 a1 = load_frag_a(A, wm + 16, lda, k0, lane);
    // Hoist ALL B-fragment LDS reads so they issue as one ds clause with a single
    // s_wait_dscnt, then the 8 WMMAs run back-to-back on the XDL pipe.
    const bf16* tile = &ldsB[buf][nwave * 64 * BKP_];
    bf16x16 bfr[4];
#pragma unroll
    for (int ni = 0; ni < 4; ++ni) bfr[ni] = load_frag_b_lds(tile, ni * 16, lane);
#pragma unroll
    for (int ni = 0; ni < 4; ++ni) {
      acc[0][ni] = __builtin_amdgcn_wmma_f32_16x16x32_bf16(false, a0, false, bfr[ni],
                                                           (short)0, acc[0][ni], false, false);
      acc[1][ni] = __builtin_amdgcn_wmma_f32_16x16x32_bf16(false, a1, false, bfr[ni],
                                                           (short)0, acc[1][ni], false, false);
    }
    asm volatile("s_wait_asynccnt 0" ::: "memory");
    __syncthreads();
    buf ^= 1;
  }
#undef STAGE_B

  // C/D layout: col = wn0 + ni*16 + (lane&15); row = wm + mi*16 + r + 8*(lane>=16)
  int cn = lane & 15;
  int ro = (lane >> 4) << 3;
#pragma unroll
  for (int mi = 0; mi < 2; ++mi) {
#pragma unroll
    for (int ni = 0; ni < 4; ++ni) {
      int col = wn0 + ni * 16 + cn;
      if (col >= N) continue;
      float bv = bias ? bias[col] : 0.0f;
#pragma unroll
      for (int r = 0; r < 8; ++r) {
        int row = wm + mi * 16 + ro + r;
        float v = acc[mi][ni][r] * scale + bv;
        long long ci = cbase + (long long)row * ldc + col;
        if (residual) v += residual[ci];
        if (relu) v = v > 0.0f ? v : 0.0f;
        if (outF) outF[ci] = v;
        else      outB[ci] = cvt_bf16(v);
      }
    }
  }
}

// ---------------- embedding: x[b,t,:] = tok_emb[idx[b,t],:] + pos_emb[t,:] ------------
__global__ __launch_bounds__(256) void gpt_embed(const int* __restrict__ idx,
                                                 const float* __restrict__ tok,
                                                 const float* __restrict__ pos,
                                                 float* __restrict__ x) {
  int row = blockIdx.x;              // 0..NTOK-1
  int t   = row & (T_ - 1);
  long long tb = (long long)idx[row] * E_;
  long long pb = (long long)t * E_;
  long long xb = (long long)row * E_;
  for (int j = 0; j < 4; ++j) {
    int c = threadIdx.x + j * 256;
    x[xb + c] = tok[tb + c] + pos[pb + c];
  }
}

// ---------------- row layernorm (f32 in -> bf16 out) ----------------------------------
__global__ __launch_bounds__(256) void gpt_layernorm(const float* __restrict__ x,
                                                     const float* __restrict__ g,
                                                     const float* __restrict__ b,
                                                     bf16* __restrict__ out) {
  __shared__ float r1[256], r2[256];
  int row = blockIdx.x;
  const float* xr = x + (long long)row * E_;
  float vals[4], s = 0.0f, sq = 0.0f;
  for (int j = 0; j < 4; ++j) {
    float v = xr[threadIdx.x + j * 256];
    vals[j] = v; s += v; sq += v * v;
  }
  r1[threadIdx.x] = s; r2[threadIdx.x] = sq;
  __syncthreads();
  for (int o = 128; o > 0; o >>= 1) {
    if (threadIdx.x < o) { r1[threadIdx.x] += r1[threadIdx.x + o]; r2[threadIdx.x] += r2[threadIdx.x + o]; }
    __syncthreads();
  }
  float mean = r1[0] * (1.0f / E_);
  float var  = r2[0] * (1.0f / E_) - mean * mean;
  float inv  = rsqrtf(var + EPS_);
  bf16* orow = out + (long long)row * E_;
  for (int j = 0; j < 4; ++j) {
    int c = threadIdx.x + j * 256;
    orow[c] = cvt_bf16((vals[j] - mean) * inv * g[c] + b[c]);
  }
}

// ---------------- causal softmax over att rows (f32 in -> bf16 probs) ------------------
__global__ __launch_bounds__(256) void gpt_softmax(const float* __restrict__ att,
                                                   bf16* __restrict__ P) {
  __shared__ float red[256];
  int tq = blockIdx.x, bh = blockIdx.y;
  long long base = ((long long)bh * T_ + tq) * T_;
  const float* a = att + base;
  bf16* p = P + base;
  int valid = tq + 1;
  float vals[4], mx = -3.4e38f;
  for (int j = 0; j < 4; ++j) {
    int c = threadIdx.x + j * 256;
    float v = (c < valid) ? a[c] : -3.4e38f;
    vals[j] = v; mx = fmaxf(mx, v);
  }
  red[threadIdx.x] = mx; __syncthreads();
  for (int o = 128; o > 0; o >>= 1) {
    if (threadIdx.x < o) red[threadIdx.x] = fmaxf(red[threadIdx.x], red[threadIdx.x + o]);
    __syncthreads();
  }
  float rowmax = red[0];
  __syncthreads();
  float s = 0.0f;
  for (int j = 0; j < 4; ++j) {
    int c = threadIdx.x + j * 256;
    float e = (c < valid) ? __expf(vals[j] - rowmax) : 0.0f;
    vals[j] = e; s += e;
  }
  red[threadIdx.x] = s; __syncthreads();
  for (int o = 128; o > 0; o >>= 1) {
    if (threadIdx.x < o) red[threadIdx.x] += red[threadIdx.x + o];
    __syncthreads();
  }
  float inv = 1.0f / red[0];
  for (int j = 0; j < 4; ++j) {
    int c = threadIdx.x + j * 256;
    p[c] = cvt_bf16(vals[j] * inv);
  }
}

// ---------------- f32 [K,N] -> bf16 transposed [N,K] (LDS tiled) -----------------------
__global__ __launch_bounds__(256) void gpt_tcvt(const float* __restrict__ w,
                                                bf16* __restrict__ wt, int K, int N) {
  __shared__ bf16 tile[32][33];
  int n0 = blockIdx.x * 32, k0 = blockIdx.y * 32;
  int tx = threadIdx.x, ty = threadIdx.y;   // block (32,8)
  for (int j = 0; j < 4; ++j) {
    int k = k0 + ty + j * 8;
    tile[ty + j * 8][tx] = cvt_bf16(w[(long long)k * N + n0 + tx]);
  }
  __syncthreads();
  for (int j = 0; j < 4; ++j) {
    int n = n0 + ty + j * 8;
    wt[(long long)n * K + k0 + tx] = tile[tx][ty + j * 8];
  }
}

// ---------------- v [B,T,H*D] bf16 -> vT [B,H,D,T] bf16 --------------------------------
__global__ __launch_bounds__(256) void gpt_vtrans(const bf16* __restrict__ v,
                                                  bf16* __restrict__ vT) {
  long long id = (long long)blockIdx.x * 256 + threadIdx.x;  // B*H*D*T total
  int t = (int)(id & (T_ - 1));
  long long rem = id >> 10;
  int d  = (int)(rem & (D_ - 1));
  int bh = (int)(rem >> 6);
  int hh = bh & (H_ - 1);
  int b  = bh >> 4;
  vT[id] = v[((long long)(b * T_ + t)) * E_ + hh * D_ + d];
}

// ======================================================================================
extern "C" void kernel_launch(void* const* d_in, const int* in_sizes, int n_in,
                              void* d_out, int out_size, void* d_ws, size_t ws_size,
                              hipStream_t stream) {
  (void)in_sizes; (void)n_in; (void)out_size; (void)ws_size;
  const int*   idx   = (const int*)  d_in[0];
  const float* tok   = (const float*)d_in[1];
  const float* pos   = (const float*)d_in[2];
  const float* ln1g  = (const float*)d_in[3];
  const float* ln1b  = (const float*)d_in[4];
  const float* Wq    = (const float*)d_in[5];
  const float* Wk    = (const float*)d_in[6];
  const float* Wv    = (const float*)d_in[7];
  const float* Wo    = (const float*)d_in[8];
  const float* bo    = (const float*)d_in[9];
  const float* ln2g  = (const float*)d_in[10];
  const float* ln2b  = (const float*)d_in[11];
  const float* W1    = (const float*)d_in[12];
  const float* b1    = (const float*)d_in[13];
  const float* W2    = (const float*)d_in[14];
  const float* b2    = (const float*)d_in[15];
  const float* lnfg  = (const float*)d_in[16];
  const float* lnfb  = (const float*)d_in[17];
  const float* Wlm   = (const float*)d_in[18];
  const float* blm   = (const float*)d_in[19];
  float* logits = (float*)d_out;

  // ---- workspace carve-out (~330 MB) ----
  char* pws = (char*)d_ws;
  auto take = [&](size_t bytes) { char* r = pws; pws += (bytes + 255) & ~(size_t)255; return (void*)r; };
  float* x    = (float*)take((size_t)NTOK * E_ * 4);
  bf16*  h    = (bf16*) take((size_t)NTOK * E_ * 2);
  bf16*  q    = (bf16*) take((size_t)NTOK * E_ * 2);
  bf16*  k    = (bf16*) take((size_t)NTOK * E_ * 2);
  bf16*  v    = (bf16*) take((size_t)NTOK * E_ * 2);
  bf16*  vT   = (bf16*) take((size_t)B_ * H_ * D_ * T_ * 2);
  bf16*  o    = (bf16*) take((size_t)NTOK * E_ * 2);
  bf16*  u    = (bf16*) take((size_t)NTOK * F_ * 2);
  float* att  = (float*)take((size_t)B_ * H_ * T_ * T_ * 4);
  bf16*  P    = (bf16*) take((size_t)B_ * H_ * T_ * T_ * 2);
  bf16*  wqT  = (bf16*) take((size_t)E_ * E_ * 2);
  bf16*  wkT  = (bf16*) take((size_t)E_ * E_ * 2);
  bf16*  wvT  = (bf16*) take((size_t)E_ * E_ * 2);
  bf16*  woT  = (bf16*) take((size_t)E_ * E_ * 2);
  bf16*  w1T  = (bf16*) take((size_t)E_ * F_ * 2);
  bf16*  w2T  = (bf16*) take((size_t)F_ * E_ * 2);
  bf16*  wlmT = (bf16*) take((size_t)V_ * E_ * 2);

  auto gemm = [&](const bf16* A, long long sA0, long long sA1, int lda,
                  const bf16* Bt, long long sB0, long long sB1, int ldb,
                  float* oF, bf16* oB, long long sC0, long long sC1, int ldc,
                  const float* bias, const float* res,
                  int M, int N, int K, int Z, int zmod, float scale, int relu) {
    dim3 g((M + BM_ - 1) / BM_, (N + BN_ - 1) / BN_, Z);
    gpt_wmma_gemm<<<g, 256, 0, stream>>>(A, sA0, sA1, Bt, sB0, sB1, oF, oB, sC0, sC1,
                                         lda, ldb, ldc, bias, res, M, N, K, zmod, scale, relu);
  };

  dim3 tb(32, 8);

  gpt_embed<<<NTOK, 256, 0, stream>>>(idx, tok, pos, x);
  gpt_tcvt<<<dim3(V_ / 32, E_ / 32), tb, 0, stream>>>(Wlm, wlmT, E_, V_);

  for (int i = 0; i < L_; ++i) {
    // attention block
    gpt_layernorm<<<NTOK, 256, 0, stream>>>(x, ln1g + i * E_, ln1b + i * E_, h);
    gpt_tcvt<<<dim3(E_ / 32, E_ / 32), tb, 0, stream>>>(Wq + (size_t)i * E_ * E_, wqT, E_, E_);
    gpt_tcvt<<<dim3(E_ / 32, E_ / 32), tb, 0, stream>>>(Wk + (size_t)i * E_ * E_, wkT, E_, E_);
    gpt_tcvt<<<dim3(E_ / 32, E_ / 32), tb, 0, stream>>>(Wv + (size_t)i * E_ * E_, wvT, E_, E_);
    gpt_tcvt<<<dim3(E_ / 32, E_ / 32), tb, 0, stream>>>(Wo + (size_t)i * E_ * E_, woT, E_, E_);
    gemm(h, 0, 0, E_, wqT, 0, 0, E_, nullptr, q, 0, 0, E_, nullptr, nullptr,
         NTOK, E_, E_, 1, 1, 1.0f, 0);
    gemm(h, 0, 0, E_, wkT, 0, 0, E_, nullptr, k, 0, 0, E_, nullptr, nullptr,
         NTOK, E_, E_, 1, 1, 1.0f, 0);
    gemm(h, 0, 0, E_, wvT, 0, 0, E_, nullptr, v, 0, 0, E_, nullptr, nullptr,
         NTOK, E_, E_, 1, 1, 1.0f, 0);
    gpt_vtrans<<<(B_ * H_ * D_ * T_) / 256, 256, 0, stream>>>(v, vT);
    // scores: att[b,h] = scale * q[b,:,h,:] @ k[b,:,h,:]^T   (per-z: zb=b, zh=h)
    gemm(q, (long long)T_ * E_, D_, E_,
         k, (long long)T_ * E_, D_, E_,
         att, nullptr, (long long)H_ * T_ * T_, (long long)T_ * T_, T_,
         nullptr, nullptr, T_, T_, D_, B_ * H_, H_, SCALE_, 0);
    gpt_softmax<<<dim3(T_, B_ * H_), 256, 0, stream>>>(att, P);
    // o[b,:,h,:] = P[b,h] @ v[b,:,h,:]  (vT is [bh, D, T] so PV is a Bt-GEMM)
    gemm(P, (long long)H_ * T_ * T_, (long long)T_ * T_, T_,
         vT, (long long)H_ * D_ * T_, (long long)D_ * T_, T_,
         nullptr, o, (long long)T_ * E_, D_, E_,
         nullptr, nullptr, T_, D_, T_, B_ * H_, H_, 1.0f, 0);
    // x = x + o @ Wo + bo
    gemm(o, 0, 0, E_, woT, 0, 0, E_, x, nullptr, 0, 0, E_,
         bo + i * E_, x, NTOK, E_, E_, 1, 1, 1.0f, 0);
    // MLP block
    gpt_layernorm<<<NTOK, 256, 0, stream>>>(x, ln2g + i * E_, ln2b + i * E_, h);
    gpt_tcvt<<<dim3(F_ / 32, E_ / 32), tb, 0, stream>>>(W1 + (size_t)i * E_ * F_, w1T, E_, F_);
    gpt_tcvt<<<dim3(E_ / 32, F_ / 32), tb, 0, stream>>>(W2 + (size_t)i * F_ * E_, w2T, F_, E_);
    gemm(h, 0, 0, E_, w1T, 0, 0, E_, nullptr, u, 0, 0, F_,
         b1 + i * F_, nullptr, NTOK, F_, E_, 1, 1, 1.0f, 1);
    gemm(u, 0, 0, F_, w2T, 0, 0, F_, x, nullptr, 0, 0, E_,
         b2 + i * E_, x, NTOK, E_, F_, 1, 1, 1.0f, 0);
  }

  // final LN + LM head
  gpt_layernorm<<<NTOK, 256, 0, stream>>>(x, lnfg, lnfb, h);
  gemm(h, 0, 0, E_, wlmT, 0, 0, E_, logits, nullptr, 0, 0, V_,
       blm, nullptr, NTOK, V_, E_, 1, 1, 1.0f, 0);
}
